// QueryRetriever_7490422964657
// MI455X (gfx1250) — matrix operand order repaired and checked by reading, not compiled, not run on hardware
//
#include <hip/hip_runtime.h>

#define N_MEM   500000
#define D_LAT   512
#define D_QRY   768
#define SEGSZ   128
#define NSEG    ((N_MEM + SEGSZ - 1) / SEGSZ)   // 3907
#define NCAND   32
#define KTOP    16
#define NEG_BIG (-3.402823466e38f)

typedef __attribute__((ext_vector_type(16))) __bf16          v16bf;
typedef __attribute__((ext_vector_type(8)))  float           v8f;
typedef __attribute__((ext_vector_type(8)))  unsigned short  us8;
typedef __attribute__((ext_vector_type(4)))  float           f4;

union Frag { v16bf v; us8 h[2]; unsigned short u[16]; };

__device__ __forceinline__ unsigned short f2bf(float f) {
  unsigned int u = __float_as_uint(f);
  u += 0x7fffu + ((u >> 16) & 1u);        // round-to-nearest-even
  return (unsigned short)(u >> 16);
}

__device__ __forceinline__ float wave_sum32(float v) {
  #pragma unroll
  for (int m = 16; m > 0; m >>= 1) v += __shfl_xor(v, m, 32);
  return v;
}

// ---------------------------------------------------------------- 1a: q = Wp @ query + bp
// one wave per output row; lanes split the 768-length dot (coalesced over columns)
__global__ __launch_bounds__(256) void proj_kernel(
    const float* __restrict__ Wp, const float* __restrict__ query,
    const float* __restrict__ bp, float* __restrict__ q_out) {
  int lane = threadIdx.x & 31;
  int o    = blockIdx.x * 8 + (threadIdx.x >> 5);   // 64 blocks * 8 waves = 512 rows
  const float* row = Wp + (size_t)o * D_QRY;
  float acc = 0.f;
  #pragma unroll
  for (int j = 0; j < D_QRY / 32; ++j) {
    int c = j * 32 + lane;
    acc += row[c] * query[c];
  }
  acc = wave_sum32(acc);
  if (lane == 0) q_out[o] = acc + bp[o];
}

// ---------------------------------------------------------------- 1b: qn = q / (||q|| + 1e-8)
__global__ __launch_bounds__(512) void norm_kernel(
    const float* __restrict__ q, float* __restrict__ qn) {
  __shared__ float red[512];
  __shared__ float s_inv;
  int t = threadIdx.x;
  float v = q[t];
  red[t] = v * v;
  for (int off = 256; off > 0; off >>= 1) {
    __syncthreads();
    if (t < off) red[t] += red[t + off];
  }
  __syncthreads();
  if (t == 0) s_inv = 1.f / (sqrtf(red[0]) + 1e-8f);
  __syncthreads();
  qn[t] = v * s_inv;
}

// ---------------------------------------------------------------- 2: sims[i] = dot(mem[i], qn) / (||mem[i]|| + 1e-8)
// HBM-bound stream of 1.02 GB: wave-per-row, b128 non-temporal loads, shfl reduce
__global__ __launch_bounds__(256) void sims_kernel(
    const float* __restrict__ memory, const float* __restrict__ qn,
    float* __restrict__ sims) {
  __shared__ float s_qn[D_LAT];
  for (int c = threadIdx.x; c < D_LAT; c += blockDim.x) s_qn[c] = qn[c];
  __syncthreads();

  int lane = threadIdx.x & 31;
  int gw   = blockIdx.x * 8 + (threadIdx.x >> 5);
  int nw   = gridDim.x * 8;
  for (int row = gw; row < N_MEM; row += nw) {
    const float* mrow = memory + (size_t)row * D_LAT;
    float dot = 0.f, ss = 0.f;
    #pragma unroll
    for (int j = 0; j < 4; ++j) {
      int e = (j * 32 + lane) * 4;               // 16B aligned, lanes contiguous per instr
      f4 m = __builtin_nontemporal_load((const f4*)(mrow + e));
      f4 qv = *(const f4*)(&s_qn[e]);
      f4 p = m * qv;
      dot += p[0] + p[1] + p[2] + p[3];
      f4 s2 = m * m;
      ss  += s2[0] + s2[1] + s2[2] + s2[3];
    }
    dot = wave_sum32(dot);
    ss  = wave_sum32(ss);
    if (lane == 0) sims[row] = dot / (sqrtf(ss) + 1e-8f);
  }
}

// ---------------------------------------------------------------- 3a: per-segment argmax (128-wide segments)
__global__ __launch_bounds__(SEGSZ) void segmax_kernel(
    const float* __restrict__ sims, float* __restrict__ segv, int* __restrict__ segi) {
  __shared__ float rv[SEGSZ];
  __shared__ int   ri[SEGSZ];
  int s = blockIdx.x, t = threadIdx.x;
  int g = s * SEGSZ + t;
  rv[t] = (g < N_MEM) ? sims[g] : NEG_BIG;
  ri[t] = g;
  for (int off = SEGSZ / 2; off > 0; off >>= 1) {
    __syncthreads();
    if (t < off) {
      if (rv[t + off] > rv[t] || (rv[t + off] == rv[t] && ri[t + off] < ri[t])) {
        rv[t] = rv[t + off]; ri[t] = ri[t + off];
      }
    }
  }
  __syncthreads();
  if (t == 0) { segv[s] = rv[0]; segi[s] = ri[0]; }
}

// ---------------------------------------------------------------- 3b: iterative top-32 over segment maxima (all in LDS)
__global__ __launch_bounds__(1024) void topk32_kernel(
    const float* __restrict__ segv, const int* __restrict__ segi,
    const float* __restrict__ sims, int* __restrict__ cand) {
  __shared__ float sv[NSEG];
  __shared__ int   si[NSEG];
  __shared__ float rv[1024];
  __shared__ int   ri[1024];
  __shared__ int   rs[1024];
  __shared__ int   chosen[NCAND];
  __shared__ int   bestSeg;
  int t = threadIdx.x;
  for (int s = t; s < NSEG; s += 1024) { sv[s] = segv[s]; si[s] = segi[s]; }
  __syncthreads();

  for (int j = 0; j < NCAND; ++j) {
    // phase 1: argmax over segment maxima
    float bv = NEG_BIG; int bi = 0x7fffffff, bs = 0;
    for (int s = t; s < NSEG; s += 1024) {
      float v = sv[s];
      if (v > bv || (v == bv && si[s] < bi)) { bv = v; bi = si[s]; bs = s; }
    }
    rv[t] = bv; ri[t] = bi; rs[t] = bs;
    for (int off = 512; off > 0; off >>= 1) {
      __syncthreads();
      if (t < off) {
        if (rv[t + off] > rv[t] || (rv[t + off] == rv[t] && ri[t + off] < ri[t])) {
          rv[t] = rv[t + off]; ri[t] = ri[t + off]; rs[t] = rs[t + off];
        }
      }
    }
    __syncthreads();
    if (t == 0) { chosen[j] = ri[0]; bestSeg = rs[0]; }
    __syncthreads();

    // phase 2: recompute the winning segment's max excluding already-chosen indices
    int s = bestSeg;
    float nv = NEG_BIG; int ni = 0x7fffffff;
    if (t < SEGSZ) {
      int g = s * SEGSZ + t;
      if (g < N_MEM) {
        bool used = false;
        for (int c = 0; c <= j; ++c) if (chosen[c] == g) { used = true; break; }
        if (!used) { nv = sims[g]; ni = g; }
      }
    }
    rv[t] = (t < SEGSZ) ? nv : NEG_BIG;
    ri[t] = (t < SEGSZ) ? ni : 0x7fffffff;
    for (int off = SEGSZ / 2; off > 0; off >>= 1) {
      __syncthreads();
      if (t < off) {
        if (rv[t + off] > rv[t] || (rv[t + off] == rv[t] && ri[t + off] < ri[t])) {
          rv[t] = rv[t + off]; ri[t] = ri[t + off];
        }
      }
    }
    __syncthreads();
    if (t == 0) { sv[s] = rv[0]; si[s] = ri[0]; }
    __syncthreads();
  }
  if (t < NCAND) cand[t] = chosen[t];
}

// ---------------------------------------------------------------- 4a: W1 f32 -> bf16 copy
__global__ __launch_bounds__(256) void w1cvt_kernel(
    const float* __restrict__ W1, unsigned short* __restrict__ w1bf) {
  int e = blockIdx.x * 256 + threadIdx.x;
  if (e < D_LAT * 2 * D_LAT) w1bf[e] = f2bf(W1[e]);
}

// ---------------------------------------------------------------- 4b: bf16 WMMA rerank MLP + fused ReLU/W2 + final top-16
// M=32, N=512, K=1024; 64 tiles of 16x16, 8 waves, 32 k-steps of v_wmma_f32_16x16x32_bf16
__global__ __launch_bounds__(256) void rerank_kernel(
    const float* __restrict__ memory, const float* __restrict__ q,
    const unsigned short* __restrict__ w1bf, const float* __restrict__ b1,
    const float* __restrict__ W2, const float* __restrict__ b2,
    const int* __restrict__ cand, float* __restrict__ out) {
  __shared__ unsigned short s_qbf[D_LAT];
  __shared__ unsigned short s_cbf[NCAND][D_LAT];
  __shared__ float s_scores[NCAND];
  int tid = threadIdx.x;

  for (int c = tid; c < D_LAT; c += 256) s_qbf[c] = f2bf(q[c]);
  for (int e = tid; e < NCAND * D_LAT; e += 256) {
    int i = e >> 9, c = e & (D_LAT - 1);
    s_cbf[i][c] = f2bf(memory[(size_t)cand[i] * D_LAT + c]);
  }
  if (tid < NCAND) s_scores[tid] = b2[0];
  __syncthreads();

  int lane = tid & 31;
  int wave = tid >> 5;
  int nl   = lane & 15;
  int hi   = lane >> 4;                       // lane half: 0 or 1

  for (int tile = wave; tile < 64; tile += 8) {
    int mt = tile & 1;                        // candidate-row tile (0..1)
    int nt = tile >> 1;                       // output-channel tile (0..31)
    int mrow = mt * 16 + nl;                  // A row for this lane
    int ncol = nt * 16 + nl;                  // B column = output channel
    v8f acc = {0.f, 0.f, 0.f, 0.f, 0.f, 0.f, 0.f, 0.f};

    for (int kt = 0; kt < 32; ++kt) {
      int kOff = kt * 32;
      Frag a, b;
      // A fragment (16x32 bf16): v<8 -> K = kOff + 8*hi + v ; v>=8 -> K = kOff + 16 + 8*hi + (v-8)
      const unsigned short* asrc =
          (kOff < D_LAT) ? &s_qbf[kOff] : &s_cbf[mrow][kOff - D_LAT];
      a.h[0] = *(const us8*)(asrc + 8 * hi);
      a.h[1] = *(const us8*)(asrc + 16 + 8 * hi);
      // B fragment (32x16 bf16): v<8 -> K = kOff + 8*hi + v ; v>=8 -> K = kOff + 16 + 8*hi + (v-8), in W1 row `ncol`
      const unsigned short* bsrc = &w1bf[(size_t)ncol * (2 * D_LAT) + kOff + 8 * hi];
      b.h[0] = *(const us8*)(bsrc);
      b.h[1] = *(const us8*)(bsrc + 16);

      acc = __builtin_amdgcn_wmma_f32_16x16x32_bf16(
          /*neg_a=*/false, a.v, /*neg_b=*/false, b.v,
          /*c_mod=*/(short)0, acc, /*reuse_a=*/false, /*reuse_b=*/false);
    }

    float w2n = W2[ncol];
    float bn  = b1[ncol];
    #pragma unroll
    for (int r = 0; r < 8; ++r) {
      int m = mt * 16 + r + 8 * hi;           // D layout: VGPR r -> M = r (+8 for hi half)
      float h = acc[r] + bn;
      h = h > 0.f ? h : 0.f;
      atomicAdd(&s_scores[m], h * w2n);       // ds_add_f32
    }
  }
  __syncthreads();

  if (tid == 0) {
    bool used[NCAND];
    float sc[NCAND];
    for (int i = 0; i < NCAND; ++i) { used[i] = false; sc[i] = s_scores[i]; }
    for (int j = 0; j < KTOP; ++j) {
      int bi = 0; float bv = NEG_BIG;
      for (int i = 0; i < NCAND; ++i)
        if (!used[i] && sc[i] > bv) { bv = sc[i]; bi = i; }
      used[bi] = true;
      out[j]        = bv;                     // top scores, descending
      out[KTOP + j] = (float)cand[bi];        // selected memory indices
    }
  }
}

// ---------------------------------------------------------------- host
extern "C" void kernel_launch(void* const* d_in, const int* in_sizes, int n_in,
                              void* d_out, int out_size, void* d_ws, size_t ws_size,
                              hipStream_t stream) {
  const float* query  = (const float*)d_in[0];
  const float* memory = (const float*)d_in[1];
  const float* Wp     = (const float*)d_in[2];
  const float* bp     = (const float*)d_in[3];
  const float* W1     = (const float*)d_in[4];
  const float* b1     = (const float*)d_in[5];
  const float* W2     = (const float*)d_in[6];
  const float* b2     = (const float*)d_in[7];
  float* out = (float*)d_out;

  char* ws = (char*)d_ws;
  float* ws_q    = (float*)(ws + 0);                 // 512 f32
  float* ws_qn   = (float*)(ws + 2048);              // 512 f32
  float* ws_sims = (float*)(ws + 4096);              // 500000 f32
  float* ws_segv = (float*)(ws + 2004096);           // NSEG f32
  int*   ws_segi = (int*)  (ws + 2019728);           // NSEG i32
  int*   ws_cand = (int*)  (ws + 2035360);           // 32 i32
  unsigned short* ws_w1bf = (unsigned short*)(ws + 2035520);  // 512*1024 bf16

  proj_kernel  <<<64,   256, 0, stream>>>(Wp, query, bp, ws_q);
  norm_kernel  <<<1,    512, 0, stream>>>(ws_q, ws_qn);
  sims_kernel  <<<4096, 256, 0, stream>>>(memory, ws_qn, ws_sims);
  segmax_kernel<<<NSEG, SEGSZ, 0, stream>>>(ws_sims, ws_segv, ws_segi);
  topk32_kernel<<<1,    1024, 0, stream>>>(ws_segv, ws_segi, ws_sims, ws_cand);
  w1cvt_kernel <<<(D_LAT * 2 * D_LAT + 255) / 256, 256, 0, stream>>>(W1, ws_w1bf);
  rerank_kernel<<<1,    256, 0, stream>>>(memory, ws_q, ws_w1bf, b1, W2, b2, ws_cand, out);
}
